// RotaryEncoderLayer_14139032338446
// MI455X (gfx1250) — compile-verified, matrix operand order checked by reading, not run on hardware
//
#include <hip/hip_runtime.h>

typedef unsigned short u16;
typedef __attribute__((ext_vector_type(16))) __bf16 v16bf;
typedef __attribute__((ext_vector_type(8)))  float  v8f;
typedef __attribute__((ext_vector_type(4)))  unsigned v4u;
typedef __attribute__((ext_vector_type(8)))  int      v8i;
typedef __attribute__((ext_vector_type(4)))  int      v4i;

union FragBF { v16bf v; uint4 q[2]; };

__device__ __forceinline__ u16 f2bf(float f) {
    unsigned u = __float_as_uint(f);
    u += 0x7FFFu + ((u >> 16) & 1u);
    return (u16)(u >> 16);
}

__device__ __forceinline__ float halfmax(float v) {
    #pragma unroll
    for (int m = 1; m < 16; m <<= 1) v = fmaxf(v, __shfl_xor(v, m, 32));
    return v;
}
__device__ __forceinline__ float halfsum(float v) {
    #pragma unroll
    for (int m = 1; m < 16; m <<= 1) v += __shfl_xor(v, m, 32);
    return v;
}

// ---------------------------------------------------------------------------
// TDM: DMA a [rows x 32] bf16 tile (row length = rowLen elems) from global
// memory into LDS. One instruction per tile, tracked with TENSORcnt.
// D# layout per CDNA5 ISA ch.8: group0 = count/lds_addr/global_addr/type,
// group1 = data_size / tensor dims / tile dims / strides.
// ---------------------------------------------------------------------------
__device__ __forceinline__ void tdm_load_tile(const u16* gsrc, unsigned ldsAddr,
                                              int tileRows, int rowLen, int totRows)
{
    unsigned long long ga = (unsigned long long)(const void*)gsrc;
    v4u g0;
    g0.x = 1u;                                           // count=1, user mode
    g0.y = ldsAddr;                                      // LDS byte address
    g0.z = (unsigned)ga;                                 // global_addr[31:0]
    g0.w = (unsigned)((ga >> 32) & 0x01FFFFFFu) | (2u << 30);  // addr[56:32] | type=2

    unsigned rl = (unsigned)rowLen, tr = (unsigned)totRows;
    v8i g1;
    g1[0] = (int)(1u << 16);                             // data_size = 2 bytes
    g1[1] = (int)((rl & 0xFFFFu) << 16);                 // tensor_dim0[15:0]
    g1[2] = (int)((rl >> 16) | ((tr & 0xFFFFu) << 16));  // dim0[31:16] | dim1[15:0]
    g1[3] = (int)((tr >> 16) | (32u << 16));             // dim1[31:16] | tile_dim0=32
    g1[4] = tileRows;                                    // tile_dim1 (tile_dim2=0)
    g1[5] = (int)rl;                                     // tensor_dim0_stride[31:0]
    g1[6] = 0;                                           // stride0 hi | stride1 lo
    g1[7] = 0;                                           // stride1 hi

    v4i z4 = {0, 0, 0, 0};
#if defined(__clang_major__) && (__clang_major__ >= 23)
    v8i z8 = {0, 0, 0, 0, 0, 0, 0, 0};
    __builtin_amdgcn_tensor_load_to_lds(g0, g1, z4, z4, z8, 0);
#else
    __builtin_amdgcn_tensor_load_to_lds(g0, g1, z4, z4, 0);
#endif
}

// ---------------------------------------------------------------------------
// fp32 -> bf16 conversion (weights / activations)
// ---------------------------------------------------------------------------
__global__ void k_f32_to_bf16(const float* __restrict__ x, u16* __restrict__ y, int n) {
    int i = blockIdx.x * blockDim.x + threadIdx.x;
    if (i < n) y[i] = f2bf(x[i]);
}

// ---------------------------------------------------------------------------
// bf16 GEMM: C[M,N] = A[M,K] @ W[N,K]^T + bias, optional ReLU, f32+bf16 out.
// 256 threads = 8 waves; block tile 128x128x32; wave tile 32x64 (2x4 wmma).
// A/B tiles staged by the Tensor Data Mover, double buffered: wave 0 issues
// TDM for tile k+1 into the alternate buffer while all waves run WMMAs on
// tile k; sync = s_wait_tensorcnt(0) + workgroup barrier.
// ---------------------------------------------------------------------------
#define BM 128
#define BN 128
#define BKK 32

__global__ __launch_bounds__(256) void k_gemm_bf16(
    const u16* __restrict__ A, const u16* __restrict__ W,
    const float* __restrict__ bias,
    float* __restrict__ Cf, u16* __restrict__ Cbf,
    int M, int N, int K, int relu)
{
    __shared__ __align__(16) u16 As[2][BM * BKK];
    __shared__ __align__(16) u16 Bs[2][BN * BKK];

    const int tid  = threadIdx.x;
    const int lane = tid & 31;
    const int w    = tid >> 5;
    const int wr   = w & 3;    // wave row (M)
    const int wc   = w >> 2;   // wave col (N)
    const int m0   = blockIdx.y * BM;
    const int n0   = blockIdx.x * BN;
    const int rlo  = lane & 15;
    const int hsel = lane >> 4;

    v8f acc[2][4];
    #pragma unroll
    for (int i = 0; i < 2; i++)
        #pragma unroll
        for (int j = 0; j < 4; j++)
            #pragma unroll
            for (int e = 0; e < 8; e++) acc[i][j][e] = 0.0f;

    const unsigned ldsA0 = (unsigned)(unsigned long long)(void*)&As[0][0];
    const unsigned ldsA1 = (unsigned)(unsigned long long)(void*)&As[1][0];
    const unsigned ldsB0 = (unsigned)(unsigned long long)(void*)&Bs[0][0];
    const unsigned ldsB1 = (unsigned)(unsigned long long)(void*)&Bs[1][0];

    // prologue: stage k0 = 0 into buffer 0
    if (w == 0) {
        tdm_load_tile(A + (size_t)m0 * K, ldsA0, BM, K, M);
        tdm_load_tile(W + (size_t)n0 * K, ldsB0, BN, K, N);
    }

    int buf = 0;
    for (int k0 = 0; k0 < K; k0 += BKK) {
        if (w == 0) __builtin_amdgcn_s_wait_tensorcnt(0);
        __syncthreads();   // current buf staged; previous buf fully consumed

        if (w == 0 && (k0 + BKK) < K) {
            int kn = k0 + BKK;
            tdm_load_tile(A + (size_t)m0 * K + kn, buf ? ldsA0 : ldsA1, BM, K, M);
            tdm_load_tile(W + (size_t)n0 * K + kn, buf ? ldsB0 : ldsB1, BN, K, N);
        }

        FragBF afr[2], bfr[4];
        #pragma unroll
        for (int mi = 0; mi < 2; mi++) {
            int row = wr * 32 + mi * 16 + rlo;
            afr[mi].q[0] = *(const uint4*)&As[buf][row * BKK + hsel * 8];
            afr[mi].q[1] = *(const uint4*)&As[buf][row * BKK + 16 + hsel * 8];
        }
        #pragma unroll
        for (int ni = 0; ni < 4; ni++) {
            int row = wc * 64 + ni * 16 + rlo;
            bfr[ni].q[0] = *(const uint4*)&Bs[buf][row * BKK + hsel * 8];
            bfr[ni].q[1] = *(const uint4*)&Bs[buf][row * BKK + 16 + hsel * 8];
        }
        #pragma unroll
        for (int mi = 0; mi < 2; mi++)
            #pragma unroll
            for (int ni = 0; ni < 4; ni++)
                acc[mi][ni] = __builtin_amdgcn_wmma_f32_16x16x32_bf16(
                    false, afr[mi].v, false, bfr[ni].v,
                    (short)0, acc[mi][ni], false, false);

        buf ^= 1;
    }

    #pragma unroll
    for (int mi = 0; mi < 2; mi++)
        #pragma unroll
        for (int ni = 0; ni < 4; ni++) {
            int gn = n0 + wc * 64 + ni * 16 + rlo;
            float bv = bias ? bias[gn] : 0.0f;
            #pragma unroll
            for (int r = 0; r < 8; r++) {
                int gm = m0 + wr * 32 + mi * 16 + hsel * 8 + r;
                float v = acc[mi][ni][r] + bv;
                if (relu) v = fmaxf(v, 0.0f);
                size_t idx = (size_t)gm * N + gn;
                if (Cf)  Cf[idx]  = v;
                if (Cbf) Cbf[idx] = f2bf(v);
            }
        }
}

// ---------------------------------------------------------------------------
// RoPE on qkv (f32 [S,B,3D]) -> q_bf, k_bf, v_bf (bf16 [S,B,D])
// ---------------------------------------------------------------------------
__global__ void k_rope(const float* __restrict__ qkv,
                       u16* __restrict__ qb, u16* __restrict__ kb, u16* __restrict__ vb,
                       int S, int B, int D)
{
    int idx = blockIdx.x * blockDim.x + threadIdx.x;
    int total = S * B * D;
    if (idx >= total) return;
    int d  = idx % D;
    int sb = idx / D;
    int s  = sb / B;
    int t  = d & 63;          // position within head
    int j  = t & 31;          // frequency index
    int hb = d - t;           // head base within D

    float inv = __powf(10000.0f, -(2.0f * (float)j) / 64.0f);
    float ang = (float)s * inv;
    float sn, c;
    __sincosf(ang, &sn, &c);

    size_t base = (size_t)sb * (3 * D);
    float xq = qkv[base + d];
    float xk = qkv[base + D + d];
    float rq, rk;
    if (t < 32) {
        rq = -qkv[base + hb + 2 * t + 1];
        rk = -qkv[base + D + hb + 2 * t + 1];
    } else {
        rq = qkv[base + hb + 2 * (t - 32)];
        rk = qkv[base + D + hb + 2 * (t - 32)];
    }
    qb[idx] = f2bf(xq * c + rq * sn);
    kb[idx] = f2bf(xk * c + rk * sn);
    vb[idx] = f2bf(qkv[base + 2 * D + d]);
}

// ---------------------------------------------------------------------------
// repack f32 [S,B,H*64] -> bf16 [B,H,S,64]
// ---------------------------------------------------------------------------
__global__ void k_repack(const float* __restrict__ x, u16* __restrict__ y,
                         int S, int B, int H)
{
    int idx = blockIdx.x * blockDim.x + threadIdx.x;
    int total = S * B * H * 64;
    if (idx >= total) return;
    int t  = idx & 63;
    int d  = idx % (H * 64);
    int h  = d >> 6;
    int sb = idx / (H * 64);
    int b  = sb % B;
    int s  = sb / B;
    y[(((size_t)b * H + h) * S + s) * 64 + t] = f2bf(x[idx]);
}

// ---------------------------------------------------------------------------
// Flash attention: one wave per (b, h, 16-query tile).
// q,k,v: bf16 [B,H,S,64]; out: ctx f32 + bf16 in [S,B,D] layout.
// ---------------------------------------------------------------------------
__global__ __launch_bounds__(256) void k_attention(
    const u16* __restrict__ Q, const u16* __restrict__ Kt, const u16* __restrict__ V,
    float* __restrict__ ctxF, u16* __restrict__ ctxBf,
    int Bsz, int Hn, int S, float scale)
{
    __shared__ __align__(16) u16 vT[8][64 * 32];   // per-wave V^T tile [hd][key]
    __shared__ __align__(16) u16 pS[8][16 * 32];   // per-wave P tile   [q][key]

    const int tid  = threadIdx.x;
    const int lane = tid & 31;
    const int w    = tid >> 5;
    const int rlo  = lane & 15;
    const int hsel = lane >> 4;

    const int gw  = blockIdx.x * 8 + w;
    const int nqt = S >> 4;
    const int qt  = gw % nqt;
    const int h   = (gw / nqt) % Hn;
    const int b   = gw / (nqt * Hn);

    const size_t headOff = ((size_t)b * Hn + h) * (size_t)S * 64;

    FragBF qa[2];
    {
        const u16* qp = Q + headOff + (size_t)(qt * 16 + rlo) * 64;
        qa[0].q[0] = *(const uint4*)(qp + hsel * 8);
        qa[0].q[1] = *(const uint4*)(qp + 16 + hsel * 8);
        qa[1].q[0] = *(const uint4*)(qp + 32 + hsel * 8);
        qa[1].q[1] = *(const uint4*)(qp + 48 + hsel * 8);
    }

    float mrow[8], lrow[8];
    v8f o[4];
    #pragma unroll
    for (int r = 0; r < 8; r++) { mrow[r] = -3.0e38f; lrow[r] = 0.0f; }
    #pragma unroll
    for (int t = 0; t < 4; t++)
        #pragma unroll
        for (int r = 0; r < 8; r++) o[t][r] = 0.0f;

    for (int kb = 0; kb < S; kb += 32) {
        // ---- scores: q (16x64) @ k^T (64x32) ----
        v8f s0, s1;
        #pragma unroll
        for (int j = 0; j < 2; j++) {
            FragBF kf0, kf1;
            const u16* kp = Kt + headOff + (size_t)(kb + j * 16 + rlo) * 64;
            kf0.q[0] = *(const uint4*)(kp + hsel * 8);
            kf0.q[1] = *(const uint4*)(kp + 16 + hsel * 8);
            kf1.q[0] = *(const uint4*)(kp + 32 + hsel * 8);
            kf1.q[1] = *(const uint4*)(kp + 48 + hsel * 8);
            v8f s;
            #pragma unroll
            for (int e = 0; e < 8; e++) s[e] = 0.0f;
            s = __builtin_amdgcn_wmma_f32_16x16x32_bf16(false, qa[0].v, false, kf0.v,
                                                        (short)0, s, false, false);
            s = __builtin_amdgcn_wmma_f32_16x16x32_bf16(false, qa[1].v, false, kf1.v,
                                                        (short)0, s, false, false);
            if (j == 0) s0 = s; else s1 = s;
        }

        // ---- stage V tile transposed into LDS: vT[hd][key] ----
        {
            union { uint4 q[8]; u16 u[64]; } vrow;
            const u16* vp = V + headOff + (size_t)(kb + lane) * 64;
            #pragma unroll
            for (int i = 0; i < 8; i++) vrow.q[i] = *(const uint4*)(vp + i * 8);
            #pragma unroll
            for (int d = 0; d < 64; d++) vT[w][d * 32 + lane] = vrow.u[d];
        }

        // ---- online softmax update (row stats in registers) ----
        #pragma unroll
        for (int r = 0; r < 8; r++) {
            float a0 = s0[r] * scale, a1 = s1[r] * scale;
            float mx = halfmax(fmaxf(a0, a1));
            float mnew = fmaxf(mrow[r], mx);
            float corr = __expf(mrow[r] - mnew);
            float p0 = __expf(a0 - mnew);
            float p1 = __expf(a1 - mnew);
            float rs = halfsum(p0 + p1);
            lrow[r] = lrow[r] * corr + rs;
            mrow[r] = mnew;
            #pragma unroll
            for (int t = 0; t < 4; t++) o[t][r] *= corr;
            int prw = hsel * 8 + r;                 // query row in D-layout
            pS[w][prw * 32 + rlo]      = f2bf(p0);
            pS[w][prw * 32 + 16 + rlo] = f2bf(p1);
        }

        // wave-local LDS RAW fence (CDNA5 split dependency counter)
        asm volatile("s_wait_dscnt 0x0" ::: "memory");

        // ---- p (16x32 keys) @ v (32 keys x 64 hd) ----
        FragBF pa;
        pa.q[0] = *(const uint4*)&pS[w][rlo * 32 + hsel * 8];
        pa.q[1] = *(const uint4*)&pS[w][rlo * 32 + 16 + hsel * 8];
        #pragma unroll
        for (int t = 0; t < 4; t++) {
            FragBF vf;
            int n = t * 16 + rlo;
            vf.q[0] = *(const uint4*)&vT[w][n * 32 + hsel * 8];
            vf.q[1] = *(const uint4*)&vT[w][n * 32 + 16 + hsel * 8];
            o[t] = __builtin_amdgcn_wmma_f32_16x16x32_bf16(false, pa.v, false, vf.v,
                                                           (short)0, o[t], false, false);
        }
    }

    // ---- epilogue: ctx[s, b, h*64 + n] = o / l ----
    #pragma unroll
    for (int t = 0; t < 4; t++)
        #pragma unroll
        for (int r = 0; r < 8; r++) {
            int grow = hsel * 8 + r;
            int s = qt * 16 + grow;
            float v = o[t][r] / lrow[r];
            size_t idx = ((size_t)s * Bsz + b) * (size_t)(Hn * 64) + h * 64 + t * 16 + rlo;
            ctxF[idx]  = v;
            ctxBf[idx] = f2bf(v);
        }
}

// ---------------------------------------------------------------------------
// out = LayerNorm(a + bres) * g + beta ; optional bf16 copy. D == 1024.
// ---------------------------------------------------------------------------
__global__ __launch_bounds__(256) void k_add_ln(
    const float* __restrict__ a, const float* __restrict__ bres,
    const float* __restrict__ g, const float* __restrict__ beta,
    float* __restrict__ out, u16* __restrict__ outbf, int D)
{
    __shared__ float red[256];
    const int row = blockIdx.x;
    const int tid = threadIdx.x;
    const float* ap = a + (size_t)row * D;
    const float* bp = bres ? bres + (size_t)row * D : nullptr;

    float x[4];
    float s = 0.0f;
    #pragma unroll
    for (int i = 0; i < 4; i++) {
        int c = tid + i * 256;
        float v = ap[c] + (bp ? bp[c] : 0.0f);
        x[i] = v; s += v;
    }
    red[tid] = s; __syncthreads();
    for (int off = 128; off > 0; off >>= 1) {
        if (tid < off) red[tid] += red[tid + off];
        __syncthreads();
    }
    float mu = red[0] / (float)D;
    __syncthreads();

    float s2 = 0.0f;
    #pragma unroll
    for (int i = 0; i < 4; i++) { float dd = x[i] - mu; s2 += dd * dd; }
    red[tid] = s2; __syncthreads();
    for (int off = 128; off > 0; off >>= 1) {
        if (tid < off) red[tid] += red[tid + off];
        __syncthreads();
    }
    float rs = rsqrtf(red[0] / (float)D + 1e-5f);

    #pragma unroll
    for (int i = 0; i < 4; i++) {
        int c = tid + i * 256;
        float v = (x[i] - mu) * rs * g[c] + beta[c];
        size_t idx = (size_t)row * D + c;
        out[idx] = v;
        if (outbf) outbf[idx] = f2bf(v);
    }
}

// ---------------------------------------------------------------------------
// host: full layer pipeline
// ---------------------------------------------------------------------------
extern "C" void kernel_launch(void* const* d_in, const int* in_sizes, int n_in,
                              void* d_out, int out_size, void* d_ws, size_t ws_size,
                              hipStream_t stream) {
    const float* src   = (const float*)d_in[0];
    const float* in_w  = (const float*)d_in[1];
    const float* in_b  = (const float*)d_in[2];
    const float* out_w = (const float*)d_in[3];
    const float* out_b = (const float*)d_in[4];
    const float* w1    = (const float*)d_in[5];
    const float* b1    = (const float*)d_in[6];
    const float* w2    = (const float*)d_in[7];
    const float* b2    = (const float*)d_in[8];
    const float* ln1g  = (const float*)d_in[9];
    const float* ln1b  = (const float*)d_in[10];
    const float* ln2g  = (const float*)d_in[11];
    const float* ln2b  = (const float*)d_in[12];

    const int S = 1024, B = 4, D = 1024, FF = 4096, H = 16;
    const int M = S * B;

    char*  base = (char*)d_ws;
    size_t off  = 0;
    auto alloc = [&](size_t bytes) -> void* {
        void* r = base + off;
        off = (off + bytes + 255) & ~(size_t)255;
        return r;
    };

    u16*   wqkv_bf = (u16*)  alloc((size_t)3 * D * D * 2);
    u16*   wo_bf   = (u16*)  alloc((size_t)D * D * 2);
    u16*   w1_bf   = (u16*)  alloc((size_t)FF * D * 2);
    u16*   w2_bf   = (u16*)  alloc((size_t)D * FF * 2);
    u16*   src_bf  = (u16*)  alloc((size_t)M * D * 2);
    float* qkv_f   = (float*)alloc((size_t)M * 3 * D * 4);
    u16*   q_bf    = (u16*)  alloc((size_t)M * D * 2);
    u16*   k_bf    = (u16*)  alloc((size_t)M * D * 2);
    u16*   v_bf    = (u16*)  alloc((size_t)M * D * 2);
    float* q2_f    = (float*)alloc((size_t)M * D * 4);
    float* k2_f    = (float*)alloc((size_t)M * D * 4);
    float* v2_f    = (float*)alloc((size_t)M * D * 4);
    u16*   qa_bf   = (u16*)  alloc((size_t)M * D * 2);
    u16*   ka_bf   = (u16*)  alloc((size_t)M * D * 2);
    u16*   va_bf   = (u16*)  alloc((size_t)M * D * 2);
    float* ctx_f   = (float*)alloc((size_t)M * D * 4);
    u16*   ctx_bf  = (u16*)  alloc((size_t)M * D * 2);
    float* attn_f  = (float*)alloc((size_t)M * D * 4);
    float* x1_f    = (float*)alloc((size_t)M * D * 4);
    u16*   x1_bf   = (u16*)  alloc((size_t)M * D * 2);
    float* h_f     = (float*)alloc((size_t)M * FF * 4);
    u16*   h_bf    = (u16*)  alloc((size_t)M * FF * 2);
    float* ff_f    = (float*)alloc((size_t)M * D * 4);

    auto cvt = [&](const float* x, u16* y, int n) {
        k_f32_to_bf16<<<(n + 255) / 256, 256, 0, stream>>>(x, y, n);
    };

    // weight + activation conversions
    cvt(in_w,  wqkv_bf, 3 * D * D);
    cvt(out_w, wo_bf,   D * D);
    cvt(w1,    w1_bf,   FF * D);
    cvt(w2,    w2_bf,   D * FF);
    cvt(src,   src_bf,  M * D);

    // 1) qkv = src @ in_proj_w^T + in_proj_b
    k_gemm_bf16<<<dim3(3 * D / BN, M / BM), 256, 0, stream>>>(
        src_bf, wqkv_bf, in_b, qkv_f, nullptr, M, 3 * D, D, 0);

    // 2) RoPE on q,k; pass-through v
    k_rope<<<(M * D + 255) / 256, 256, 0, stream>>>(qkv_f, q_bf, k_bf, v_bf, S, B, D);

    // 3) secondary projections q2/k2/v2 (same weights, per reference)
    k_gemm_bf16<<<dim3(D / BN, M / BM), 256, 0, stream>>>(
        q_bf, wqkv_bf,             in_b,         q2_f, nullptr, M, D, D, 0);
    k_gemm_bf16<<<dim3(D / BN, M / BM), 256, 0, stream>>>(
        k_bf, wqkv_bf + (size_t)D * D,     in_b + D,     k2_f, nullptr, M, D, D, 0);
    k_gemm_bf16<<<dim3(D / BN, M / BM), 256, 0, stream>>>(
        v_bf, wqkv_bf + (size_t)2 * D * D, in_b + 2 * D, v2_f, nullptr, M, D, D, 0);

    // 4) repack to [B,H,S,hd] bf16 for attention
    k_repack<<<(M * D + 255) / 256, 256, 0, stream>>>(q2_f, qa_bf, S, B, H);
    k_repack<<<(M * D + 255) / 256, 256, 0, stream>>>(k2_f, ka_bf, S, B, H);
    k_repack<<<(M * D + 255) / 256, 256, 0, stream>>>(v2_f, va_bf, S, B, H);

    // 5) flash attention (scale = 1/sqrt(64))
    {
        int waves = B * H * (S / 16);
        k_attention<<<waves / 8, 256, 0, stream>>>(
            qa_bf, ka_bf, va_bf, ctx_f, ctx_bf, B, H, S, 0.125f);
    }

    // 6) out projection
    k_gemm_bf16<<<dim3(D / BN, M / BM), 256, 0, stream>>>(
        ctx_bf, wo_bf, out_b, attn_f, nullptr, M, D, D, 0);

    // 7) x = LN(src + attn_out)
    k_add_ln<<<M, 256, 0, stream>>>(src, attn_f, ln1g, ln1b, x1_f, x1_bf, D);

    // 8) FFN: h = relu(x @ w1^T + b1), ff = h @ w2^T + b2
    k_gemm_bf16<<<dim3(FF / BN, M / BM), 256, 0, stream>>>(
        x1_bf, w1_bf, b1, h_f, h_bf, M, FF, D, 1);
    k_gemm_bf16<<<dim3(D / BN, M / BM), 256, 0, stream>>>(
        h_bf, w2_bf, b2, ff_f, nullptr, M, D, FF, 0);

    // 9) out = LN(x + ff)
    k_add_ln<<<M, 256, 0, stream>>>(x1_f, ff_f, ln2g, ln2b, (float*)d_out, nullptr, D);

    (void)in_sizes; (void)n_in; (void)out_size; (void)ws_size;
}
